// ARPS_46084999086287
// MI455X (gfx1250) — compile-verified, matrix operand order checked by reading, not run on hardware
//
#include <hip/hip_runtime.h>

typedef __attribute__((ext_vector_type(16))) _Float16 v16h;
typedef __attribute__((ext_vector_type(8)))  _Float16 v8h;
typedef __attribute__((ext_vector_type(8)))  float    v8f;
typedef __attribute__((ext_vector_type(4)))  int      v4i;

#define DIM 256
#define TOK 900
#define TP  912          // 57 * 16, padded token rows
#define NB  32
#define NHEAD 4
#define HDIM 64
#define LSTEP 8
#define PPRIM 12

#if __has_builtin(__builtin_amdgcn_global_load_async_to_lds_b128) && \
    __has_builtin(__builtin_amdgcn_s_wait_asynccnt)
#define HAVE_ASYNC_LDS 1
#endif

#define LDS_AS __attribute__((address_space(3)))

#ifdef HAVE_ASYNC_LDS
static __device__ __forceinline__ LDS_AS v4i* as_lds_v4i(void* p) {
  return (LDS_AS v4i*)(unsigned int)(unsigned long long)p;
}
#endif

// ---- fragment helpers -------------------------------------------------------
// A/B fragment for v_wmma_f32_16x16x32_f16: per lane, 16 halves =
// [k = hi8..hi8+7] ++ [k = 16+hi8..16+hi8+7] of one row (contiguous memory).
__device__ __forceinline__ v16h ldfrag(const _Float16* p) {
  v8h lo = *(const v8h*)(p);
  v8h hi = *(const v8h*)(p + 16);
  return __builtin_shufflevector(lo, hi, 0,1,2,3,4,5,6,7,8,9,10,11,12,13,14,15);
}

__device__ __forceinline__ v8f wmma_f16(v16h a, v16h b, v8f c) {
  return __builtin_amdgcn_wmma_f32_16x16x32_f16(false, a, false, b, (short)0, c,
                                                false, false);
}

// ---- prep kernels -----------------------------------------------------------
__global__ void transpose_w(const float* __restrict__ W, _Float16* __restrict__ Wt) {
  int id = blockIdx.x * 256 + threadIdx.x;      // DIM*DIM threads
  int k = id % DIM, n = id / DIM;
  Wt[(long)n * DIM + k] = (_Float16)W[(long)k * DIM + n];
}

__global__ void prep_sq(const float* __restrict__ sq, _Float16* __restrict__ sqh) {
  int id = blockIdx.x * 256 + threadIdx.x;      // 16*DIM threads
  int k = id % DIM, r = id / DIM;
  sqh[id] = (_Float16)((r < LSTEP) ? sq[r * DIM + k] : 0.f);
}

__global__ void embed_kernel(const int* __restrict__ grid,
                             const float* __restrict__ embed,
                             const float* __restrict__ pos,
                             float* __restrict__ tokF, _Float16* __restrict__ tokH) {
  long id = (long)blockIdx.x * 256 + threadIdx.x;   // NB*TP*DIM threads
  int d = (int)(id % DIM);
  long bt = id / DIM;
  int t = (int)(bt % TP);
  int b = (int)(bt / TP);
  if (b >= NB) return;
  float v = 0.f;
  if (t < TOK) {
    int g = grid[b * TOK + t];
    v = embed[g * DIM + d] + pos[(long)t * DIM + d];
  }
  tokF[id] = v;
  tokH[id] = (_Float16)v;
}

// ---- WMMA GEMM, 2x2 register-blocked: each wave owns a 32x32 C tile ---------
// C[M x 256] = A[M x 256] @ Bt^T (Bt is N x K, pre-transposed weights).
// 4 WMMAs per 4 fragment loads -> 2 b128 loads per WMMA. Pad rows -> 0.
__global__ __launch_bounds__(256) void gemm_ab(
    const _Float16* __restrict__ A, long aBatch,
    const _Float16* __restrict__ Bt,
    const float* __restrict__ resid, long rBatch,
    _Float16* __restrict__ C16, long c16Batch,
    float* __restrict__ C32, long c32Batch,
    int M, int tilesM32, int rowsPad) {
  int lane = threadIdx.x & 31, w = threadIdx.x >> 5;
  int tm = blockIdx.y * 4 + (w & 3);            // 32-row tile
  int tn = blockIdx.x * 2 + (w >> 2);           // 32-col tile
  if (tm >= tilesM32) return;
  int b = blockIdx.z;
  int l15 = lane & 15, hi8 = (lane >> 4) * 8;
  int r0 = tm * 32 + l15;      if (r0 >= rowsPad) r0 = 0;   // clamp pad reads
  int r1 = tm * 32 + 16 + l15; if (r1 >= rowsPad) r1 = 0;
  int n0 = tn * 32 + l15;
  int n1 = n0 + 16;
  const _Float16* Ab = A + (long)b * aBatch;
  const _Float16* ap0 = Ab + (long)r0 * DIM + hi8;
  const _Float16* ap1 = Ab + (long)r1 * DIM + hi8;
  const _Float16* bp0 = Bt + (long)n0 * DIM + hi8;
  const _Float16* bp1 = Bt + (long)n1 * DIM + hi8;
  v8f c00 = {}, c01 = {}, c10 = {}, c11 = {};
#pragma unroll
  for (int k0 = 0; k0 < DIM; k0 += 32) {
    v16h a0 = ldfrag(ap0 + k0);
    v16h a1 = ldfrag(ap1 + k0);
    v16h b0 = ldfrag(bp0 + k0);
    v16h b1 = ldfrag(bp1 + k0);
    c00 = wmma_f16(a0, b0, c00);
    c01 = wmma_f16(a0, b1, c01);
    c10 = wmma_f16(a1, b0, c10);
    c11 = wmma_f16(a1, b1, c11);
  }
  v8f* cc[4] = { &c00, &c01, &c10, &c11 };
#pragma unroll
  for (int sm = 0; sm < 2; ++sm) {
#pragma unroll
    for (int sn = 0; sn < 2; ++sn) {
      v8f& acc = *cc[sm * 2 + sn];
      int nc = tn * 32 + sn * 16 + l15;
#pragma unroll
      for (int e = 0; e < 8; ++e) {
        int m = tm * 32 + sm * 16 + e + hi8;
        if (m >= rowsPad) continue;
        float v = acc[e];
        if (resid) v += resid[(long)b * rBatch + (long)m * DIM + nc];
        if (m >= M) v = 0.f;
        if (C32) C32[(long)b * c32Batch + (long)m * DIM + nc] = v;
        if (C16) C16[(long)b * c16Batch + (long)m * DIM + nc] = (_Float16)v;
      }
    }
  }
}

// ---- flash attention (wave32 WMMA, online softmax) --------------------------
// block = 8 waves sharing one (b, head); each wave owns one 16-row q tile.
// Scores computed transposed (St = K_chunk @ Q^T) so exp() results are already
// in A-fragment layout for the P @ V WMMA. K chunk staged to LDS once per
// block (async DMA on gfx1250 when available); V chunk staged transposed.
__global__ __launch_bounds__(256) void flash_attn(
    const _Float16* __restrict__ Q, long qBatch, int Mq, int qRows,
    const _Float16* __restrict__ K, const _Float16* __restrict__ V,
    long kvBatch, int Tkv,
    _Float16* __restrict__ O, long oBatch, float scale) {
  __shared__ _Float16 Kt[32][64];               // [key][dim] row-major chunk
  __shared__ _Float16 Vt[64][40];               // [dim][key], padded stride
  int lane = threadIdx.x & 31, w = threadIdx.x >> 5;
  int b = blockIdx.z, h = blockIdx.y;
  int l15 = lane & 15, hi8 = (lane >> 4) * 8;
  int q0 = (blockIdx.x * 8 + w) * 16;
  const _Float16* Qb = Q + (long)b * qBatch + h * HDIM;
  const _Float16* Kb = K + (long)b * kvBatch + h * HDIM;
  const _Float16* Vb = V + (long)b * kvBatch + h * HDIM;

  int qr = q0 + l15; if (qr >= qRows) qr = 0;   // clamp pad tiles in-buffer
  const _Float16* qp = Qb + (long)qr * DIM + hi8;
  v16h qa0 = ldfrag(qp);                        // kdim 0..31  (B-op of K@Q^T)
  v16h qa1 = ldfrag(qp + 32);                   // kdim 32..63

  v8f  oAcc[4] = {};
  float m_run = -1e30f, l_run = 0.f;
  int nchunks = (Tkv + 31) / 32;

  for (int ch = 0; ch < nchunks; ++ch) {
    int key0 = ch * 32;
    // V chunk (32 keys x 64 dims): prefetch to regs before the barrier
    int tt = threadIdx.x, vk = tt >> 3, vd0 = (tt & 7) * 8;
    v8h vv = {};
    if (key0 + vk < Tkv) vv = *(const v8h*)(Vb + (long)(key0 + vk) * DIM + vd0);
    __syncthreads();                            // prior-chunk LDS reads done
    // transpose-store V into LDS
#pragma unroll
    for (int j = 0; j < 8; ++j) Vt[vd0 + j][vk] = vv[j];
    // stage K chunk row-major into LDS (shared by all 8 waves)
#ifdef HAVE_ASYNC_LDS
    if (w == 0) {
#pragma unroll
      for (int j = 0; j < 8; ++j) {             // 32 lanes x 16B = 4 rows/iter
        int krow = key0 + j * 4 + (lane >> 3); if (krow >= Tkv) krow = Tkv - 1;
        const _Float16* gs = Kb + (long)krow * DIM + (lane & 7) * 8;
        void* ls = &Kt[j * 4 + (lane >> 3)][(lane & 7) * 8];
        __builtin_amdgcn_global_load_async_to_lds_b128((v4i*)gs, as_lds_v4i(ls),
                                                       0, 0);
      }
      __builtin_amdgcn_s_wait_asynccnt(0);
    }
#else
    {
      int krow = key0 + (tt >> 3); if (krow >= Tkv) krow = Tkv - 1;
      v8h kv = *(const v8h*)(Kb + (long)krow * DIM + (tt & 7) * 8);
      *(v8h*)&Kt[tt >> 3][(tt & 7) * 8] = kv;
    }
#endif
    __syncthreads();

    // St tiles from LDS K: lane = q-row, vgpr e = key-in-tile
    const _Float16* kls0 = (const _Float16*)&Kt[l15][hi8];
    const _Float16* kls1 = (const _Float16*)&Kt[16 + l15][hi8];
    v8f st0 = {}, st1 = {};
    st0 = wmma_f16(ldfrag(kls0), qa0, st0);
    st0 = wmma_f16(ldfrag(kls0 + 32), qa1, st0);
    st1 = wmma_f16(ldfrag(kls1), qa0, st1);
    st1 = wmma_f16(ldfrag(kls1 + 32), qa1, st1);

    float s0[8], s1[8], lmax = -1e30f;
#pragma unroll
    for (int e = 0; e < 8; ++e) {
      s0[e] = (key0 + e + hi8      < Tkv) ? st0[e] * scale : -1e30f;
      s1[e] = (key0 + 16 + e + hi8 < Tkv) ? st1[e] * scale : -1e30f;
      lmax = fmaxf(lmax, fmaxf(s0[e], s1[e]));
    }
    float cmax = fmaxf(lmax, __shfl_xor(lmax, 16, 32));
    float newm = fmaxf(m_run, cmax);
    float corr = __expf(m_run - newm);
    v16h pa;                                    // == A-fragment of P, for free
    float lsum = 0.f;
#pragma unroll
    for (int e = 0; e < 8; ++e) {
      float p0 = __expf(s0[e] - newm);
      float p1 = __expf(s1[e] - newm);
      pa[e] = (_Float16)p0;
      pa[8 + e] = (_Float16)p1;
      lsum += p0 + p1;
    }
    l_run = l_run * corr + (lsum + __shfl_xor(lsum, 16, 32));
    m_run = newm;
    // rescale running output (O C-layout: lane = dim, row = e + hi8)
#pragma unroll
    for (int e = 0; e < 8; ++e) {
      float f = __shfl(corr, e + hi8, 32);
      oAcc[0][e] *= f; oAcc[1][e] *= f; oAcc[2][e] *= f; oAcc[3][e] *= f;
    }
    // P @ V : B-fragments from transposed-V LDS tile
#pragma unroll
    for (int dt = 0; dt < 4; ++dt) {
      const _Float16* vp = (const _Float16*)&Vt[dt * 16 + l15][hi8];
      oAcc[dt] = wmma_f16(pa, ldfrag(vp), oAcc[dt]);
    }
  }
  // normalize + store
  float inv = (l_run > 0.f) ? 1.f / l_run : 0.f;
#pragma unroll
  for (int e = 0; e < 8; ++e) {
    float f = __shfl(inv, e + hi8, 32);
    int m = q0 + e + hi8;
    if (m < Mq) {
      long base = (long)b * oBatch + (long)m * DIM + h * HDIM;
#pragma unroll
      for (int dt = 0; dt < 4; ++dt)
        O[base + dt * 16 + l15] = (_Float16)(oAcc[dt][e] * f);
    }
  }
}

// ---- logits = hsteps @ w_out + b_out ---------------------------------------
__global__ void logits_kernel(const float* __restrict__ hst,
                              const float* __restrict__ w_out,
                              const float* __restrict__ b_out,
                              float* __restrict__ out) {
  int id = blockIdx.x * 64 + threadIdx.x;
  if (id >= NB * LSTEP * PPRIM) return;
  int p = id % PPRIM;
  int l = (id / PPRIM) % LSTEP;
  int b = id / (PPRIM * LSTEP);
  float s = b_out[p];
  const float* hrow = hst + ((long)b * 16 + l) * DIM;
  for (int d = 0; d < DIM; ++d) s += hrow[d] * w_out[d * PPRIM + p];
  out[id] = s;
}

// ---- program executor: flood fill -> rotate90 about anchor -> paint --------
__global__ __launch_bounds__(256) void exec_kernel(const int* __restrict__ grid,
                                                   const int* __restrict__ anchor,
                                                   float* __restrict__ out) {
  __shared__ unsigned char gg[900], same[900], bufA[900], bufB[900], rot[900];
  __shared__ int changed;
  int b = blockIdx.x, tid = threadIdx.x;
  int ar = anchor[b * 2 + 0]; ar = ar < 0 ? 0 : (ar > 29 ? 29 : ar);
  int ac = anchor[b * 2 + 1]; ac = ac < 0 ? 0 : (ac > 29 ? 29 : ac);
  for (int i = tid; i < 900; i += 256) gg[i] = (unsigned char)grid[b * 900 + i];
  __syncthreads();
  unsigned char seed = gg[ar * 30 + ac];
  for (int i = tid; i < 900; i += 256) {
    same[i] = (gg[i] == seed);
    bufA[i] = 0;
    rot[i] = 0;
  }
  __syncthreads();
  if (tid == 0) bufA[ar * 30 + ac] = 1;
  __syncthreads();
  unsigned char* cur = bufA;
  unsigned char* nxt = bufB;
  for (int it = 0; it < 900; ++it) {
    if (tid == 0) changed = 0;
    __syncthreads();
    for (int i = tid; i < 900; i += 256) {
      int r = i / 30, c = i % 30;
      unsigned char nb = cur[i];
      if (r > 0)  nb |= cur[i - 30];
      if (r < 29) nb |= cur[i + 30];
      if (c > 0)  nb |= cur[i - 1];
      if (c < 29) nb |= cur[i + 1];
      unsigned char nv = (unsigned char)(nb & same[i]);
      nxt[i] = nv;
      if (nv != cur[i]) changed = 1;
    }
    __syncthreads();
    if (!changed) break;
    unsigned char* t2 = cur; cur = nxt; nxt = t2;
    __syncthreads();
  }
  __syncthreads();
  for (int i = tid; i < 900; i += 256) {
    if (cur[i]) {
      int r = i / 30, c = i % 30;
      int nr = ar + (c - ac), nc = ac - (r - ar);
      if (nr >= 0 && nr < 30 && nc >= 0 && nc < 30) rot[nr * 30 + nc] = 1;
    }
  }
  __syncthreads();
  for (int i = tid; i < 900; i += 256)
    out[b * 900 + i] = (float)(rot[i] ? 3 : (int)gg[i]);
}

// ---- host driver ------------------------------------------------------------
extern "C" void kernel_launch(void* const* d_in, const int* in_sizes, int n_in,
                              void* d_out, int out_size, void* d_ws, size_t ws_size,
                              hipStream_t stream) {
  (void)in_sizes; (void)n_in; (void)out_size; (void)ws_size;
  const int*   gIn    = (const int*)d_in[0];
  const int*   anchor = (const int*)d_in[1];
  const float* embed  = (const float*)d_in[2];
  const float* pos    = (const float*)d_in[3];
  const float* wq     = (const float*)d_in[4];
  const float* wk     = (const float*)d_in[5];
  const float* wv     = (const float*)d_in[6];
  const float* wo     = (const float*)d_in[7];
  const float* sq     = (const float*)d_in[8];
  const float* wout   = (const float*)d_in[9];
  const float* bout   = (const float*)d_in[10];
  float* out = (float*)d_out;

  char* ws = (char*)d_ws;
  size_t off = 0;
  auto alloc = [&](size_t bytes) -> void* {
    void* p = ws + off;
    off = (off + bytes + 255) & ~(size_t)255;
    return p;
  };
  const long BT = (long)NB * TP * DIM;
  float*    tokF = (float*)alloc(BT * 4);        // tokens f32 (residual)
  _Float16* A16  = (_Float16*)alloc(BT * 2);     // tokens f16 (then tok2 f16)
  _Float16* QH   = (_Float16*)alloc(BT * 2);
  _Float16* KH   = (_Float16*)alloc(BT * 2);
  _Float16* VH   = (_Float16*)alloc(BT * 2);
  _Float16* AOH  = (_Float16*)alloc(BT * 2);     // self-attn out f16
  _Float16* WqT  = (_Float16*)alloc((size_t)DIM * DIM * 2);
  _Float16* WkT  = (_Float16*)alloc((size_t)DIM * DIM * 2);
  _Float16* WvT  = (_Float16*)alloc((size_t)DIM * DIM * 2);
  _Float16* WoT  = (_Float16*)alloc((size_t)DIM * DIM * 2);
  _Float16* SQH  = (_Float16*)alloc((size_t)16 * DIM * 2);
  _Float16* QC   = (_Float16*)alloc((size_t)16 * DIM * 2);
  _Float16* XO   = (_Float16*)alloc((size_t)NB * 16 * DIM * 2);
  float*    HST  = (float*)alloc((size_t)NB * 16 * DIM * 4);

  const long TPD = (long)TP * DIM;
  // prep
  transpose_w<<<DIM, 256, 0, stream>>>(wq, WqT);
  transpose_w<<<DIM, 256, 0, stream>>>(wk, WkT);
  transpose_w<<<DIM, 256, 0, stream>>>(wv, WvT);
  transpose_w<<<DIM, 256, 0, stream>>>(wo, WoT);
  prep_sq<<<16, 256, 0, stream>>>(sq, SQH);
  embed_kernel<<<(unsigned)(BT / 256), 256, 0, stream>>>(gIn, embed, pos, tokF, A16);

  // 29 x 32-row tiles cover 928 >= TP; 4 n-blocks x 2 n-waves x 32 = 256 cols
  dim3 gBig(4, 8, NB);
  // Q/K/V projections (f16 out, scale folded into attention)
  gemm_ab<<<gBig, 256, 0, stream>>>(A16, TPD, WqT, nullptr, 0, QH, TPD, nullptr, 0, TOK, 29, TP);
  gemm_ab<<<gBig, 256, 0, stream>>>(A16, TPD, WkT, nullptr, 0, KH, TPD, nullptr, 0, TOK, 29, TP);
  gemm_ab<<<gBig, 256, 0, stream>>>(A16, TPD, WvT, nullptr, 0, VH, TPD, nullptr, 0, TOK, 29, TP);
  // self attention over 900 pixels
  dim3 gFA(8, NHEAD, NB);
  flash_attn<<<gFA, 256, 0, stream>>>(QH, TPD, TOK, TP, KH, VH, TPD, TOK, AOH, TPD, 0.125f);
  // residual: tok2 = tok + AO @ wo  (f32 in place, f16 copy for next GEMMs)
  gemm_ab<<<gBig, 256, 0, stream>>>(AOH, TPD, WoT, tokF, TPD, A16, TPD, tokF, TPD, TOK, 29, TP);
  // K/V for cross attention from tok2
  gemm_ab<<<gBig, 256, 0, stream>>>(A16, TPD, WkT, nullptr, 0, KH, TPD, nullptr, 0, TOK, 29, TP);
  gemm_ab<<<gBig, 256, 0, stream>>>(A16, TPD, WvT, nullptr, 0, VH, TPD, nullptr, 0, TOK, 29, TP);
  // step queries: qc = step_q @ wq (batch independent)
  dim3 gOne(4, 1, 1);
  gemm_ab<<<gOne, 256, 0, stream>>>(SQH, 0, WqT, nullptr, 0, QC, 0, nullptr, 0, LSTEP, 1, 16);
  // cross attention: steps -> pixels
  dim3 gFC(1, NHEAD, NB);
  flash_attn<<<gFC, 256, 0, stream>>>(QC, 0, LSTEP, 16, KH, VH, TPD, TOK, XO,
                                      (long)16 * DIM, 0.125f);
  // hsteps = XO @ wo (f32)
  dim3 gH(4, 1, NB);
  gemm_ab<<<gH, 256, 0, stream>>>(XO, (long)16 * DIM, WoT, nullptr, 0, nullptr, 0,
                                  HST, (long)16 * DIM, LSTEP, 1, 16);
  // logits + executed grids
  logits_kernel<<<(NB * LSTEP * PPRIM + 63) / 64, 64, 0, stream>>>(HST, wout, bout, out);
  exec_kernel<<<NB, 256, 0, stream>>>(gIn, anchor, out + NB * LSTEP * PPRIM);
}
